// Conditional_Pos_Encoder_3874060501266
// MI455X (gfx1250) — matrix operand order repaired and checked by reading, not compiled, not run on hardware
//
#include <hip/hip_runtime.h>

#define N_VOXELS 100000
#define CIN      64
#define COUT     64
#define KOFF     27
#define TILE_M   128
#define Y_STRIDE 68   // padded f32 row stride to break the 64-bank pattern

// ws layout (uints): [feats_bf: N*32][wconv frags: 27*2*4*32*8][wlin frags: 2*4*32*8]
#define WCONV_FRAG_U (KOFF * 2 * 4 * 32 * 8)   // 55296 uints
#define WLIN_FRAG_U  (2 * 4 * 32 * 8)          // 2048 uints

typedef unsigned int u32;
typedef __attribute__((ext_vector_type(16))) __bf16 bf16x16;
typedef __attribute__((ext_vector_type(8)))  float  v8f;

union Frag16 {
  bf16x16 v;
  uint4   q[2];
  u32     u[8];
};

__device__ __forceinline__ u32 pack_bf16(float a, float b) {
  u32 ua = __float_as_uint(a);
  u32 ub = __float_as_uint(b);
  ua = (ua + 0x7FFFu + ((ua >> 16) & 1u)) >> 16;   // RNE f32 -> bf16
  ub = (ub + 0x7FFFu + ((ub >> 16) & 1u)) >> 16;
  return ua | (ub << 16);
}

__device__ __forceinline__ unsigned short f2bf(float a) {
  u32 ua = __float_as_uint(a);
  return (unsigned short)((ua + 0x7FFFu + ((ua >> 16) & 1u)) >> 16);
}

__device__ __forceinline__ v8f zero8() {
  v8f z;
#pragma unroll
  for (int i = 0; i < 8; ++i) z[i] = 0.0f;
  return z;
}

// ---------------- prep: feats f32 -> packed bf16 pairs ----------------------
__global__ __launch_bounds__(256) void cvt_feats_kernel(
    const float* __restrict__ f, u32* __restrict__ o, int n_pairs) {
  int i = blockIdx.x * 256 + threadIdx.x;
  if (i < n_pairs) {
    const float2 v = *(const float2*)(f + (long long)i * 2);
    o[i] = pack_bf16(v.x, v.y);
  }
}

// ---------------- prep: W_conv -> bf16 WMMA B-fragment layout ---------------
// layout: [off][s][ntile][lane][j]; lane=(h,nl): element 2j -> K=32s+16h+2j, N=16nt+nl
__global__ __launch_bounds__(256) void cvt_wconv_kernel(
    const float* __restrict__ Wc, u32* __restrict__ o) {
  int i = blockIdx.x * 256 + threadIdx.x;
  if (i < WCONV_FRAG_U) {
    int j    = i & 7;
    int lane = (i >> 3) & 31;
    int nt   = (i >> 8) & 3;
    int s    = (i >> 10) & 1;
    int off  = i >> 11;
    int h = lane >> 4, nl = lane & 15;
    int k = 32 * s + 16 * h + 2 * j;
    int n = 16 * nt + nl;
    const float* p = Wc + off * (CIN * COUT) + k * COUT + n;
    o[i] = pack_bf16(p[0], p[COUT]);
  }
}

// ---------------- prep: W_lin -> bf16 WMMA B-fragment layout ----------------
__global__ __launch_bounds__(256) void cvt_wlin_kernel(
    const float* __restrict__ Wl, u32* __restrict__ o) {
  int i = blockIdx.x * 256 + threadIdx.x;
  if (i < WLIN_FRAG_U) {
    int j    = i & 7;
    int lane = (i >> 3) & 31;
    int nt   = (i >> 8) & 3;
    int s    = (i >> 10) & 1;
    int h = lane >> 4, nl = lane & 15;
    int k = 32 * s + 16 * h + 2 * j;
    int n = 16 * nt + nl;
    o[i] = pack_bf16(Wl[k * COUT + n], Wl[(k + 1) * COUT + n]);
  }
}

// gather one offset's 128 neighbor rows (bf16) into an LDS A buffer, with
// 16B-chunk XOR swizzle (chunk ^= (row>>1)&7) to kill ds_load bank conflicts
__device__ __forceinline__ void gather_offset(
    const u32* __restrict__ feats_bf, const int* __restrict__ nb,
    int vrow_mine, int off, int wv, int lane, u32* abuf) {
  int myidx = (vrow_mine < N_VOXELS) ? nb[vrow_mine * KOFF + off] : -1;
  const int sub = lane >> 3;        // 0..3 : row within 4-row group
  const int ch  = lane & 7;         // 16B chunk within a 128B row
#pragma unroll
  for (int r = 0; r < 8; ++r) {
    int rloc = (r * 4 + sub) + wv * 32;
    int idx  = __builtin_amdgcn_ds_bpermute(((r * 4 + sub) << 2), myidx);
    uint4 d = {0u, 0u, 0u, 0u};
    if (idx >= 0) d = *(const uint4*)(feats_bf + (long long)idx * 32 + ch * 4);
    int chs = ch ^ ((rloc >> 1) & 7);
    *(uint4*)(&abuf[rloc * 32 + chs * 4]) = d;
  }
}

// ---------------- main fused kernel -----------------------------------------
__global__ __launch_bounds__(128) void sparse_conv_lin_ln_kernel(
    const u32*   __restrict__ feats_bf, // [N][32] packed bf16 pairs
    const int*   __restrict__ nb,       // [N, 27]
    const u32*   __restrict__ wcf,      // W_conv fragments
    const float* __restrict__ bc,       // [64]
    const u32*   __restrict__ wlf,      // W_lin fragments
    const float* __restrict__ bl,       // [64]
    const float* __restrict__ gamma,    // [64]
    const float* __restrict__ beta,     // [64]
    float*       __restrict__ out)      // [N, 64]
{
  __shared__ __align__(16) union {
    u32 a[2][TILE_M * 32];                         // double-buffered bf16 A (32 KB)
    struct {
      u32   x[TILE_M * 32];                        // bf16 conv output (16 KB)
      float y[TILE_M * Y_STRIDE];                  // f32 linear output (34 KB)
    } post;
  } smem;

  const int tid  = threadIdx.x;
  const int wv   = tid >> 5;        // wave id -> output-channel tile
  const int lane = tid & 31;
  const int h    = lane >> 4;
  const int nl   = lane & 15;
  const int v0   = blockIdx.x * TILE_M;
  const int swz  = nl >> 1;         // chunk swizzle term for fragment loads

  v8f acc[8];
#pragma unroll
  for (int mt = 0; mt < 8; ++mt) acc[mt] = zero8();

  const int vrow_mine = v0 + wv * 32 + lane;

  // --- prologue gather (offset 0 -> buffer 0) ---
  gather_offset(feats_bf, nb, vrow_mine, 0, wv, lane, smem.a[0]);
  __syncthreads();

  // --- conv: 27 offsets, double-buffered ---
#pragma unroll 1
  for (int off = 0; off < KOFF; ++off) {
    const int cur = off & 1;

    // issue next offset's gather first so it overlaps the WMMA sweep
    if (off + 1 < KOFF)
      gather_offset(feats_bf, nb, vrow_mine, off + 1, wv, lane,
                    smem.a[cur ^ 1]);

    // WMMA sweep over current buffer
    const u32* abuf = smem.a[cur];
#pragma unroll
    for (int s = 0; s < 2; ++s) {
      Frag16 bf;
      const u32* bp = wcf + (((off * 2 + s) * 4 + wv) * 32 + lane) * 8;
      bf.q[0] = *(const uint4*)(bp);
      bf.q[1] = *(const uint4*)(bp + 4);
      // lane-constant swizzled chunk bases; mt sweep stays immediate-offset
      const u32* ap0 = abuf + nl * 32 + (((4 * s + 2 * h)     ^ swz) * 4);
      const u32* ap1 = abuf + nl * 32 + (((4 * s + 2 * h + 1) ^ swz) * 4);
#pragma unroll
      for (int mt = 0; mt < 8; ++mt) {
        Frag16 af;
        af.q[0] = *(const uint4*)(ap0 + mt * 512);
        af.q[1] = *(const uint4*)(ap1 + mt * 512);
        acc[mt] = __builtin_amdgcn_wmma_f32_16x16x32_bf16(
            false, af.v, false, bf.v, (short)0, acc[mt], false, false);
      }
    }
    __syncthreads();
  }

  // --- bias + restage conv output as bf16 (aliases conv buffers, post-barrier)
  {
    const float bcv = bc[16 * wv + nl];
    unsigned short* sx = (unsigned short*)smem.post.x;
    const int chx = (16 * wv + nl) >> 3;     // 16B chunk of this lane's channel
    const int low = (16 * wv + nl) & 7;      // bf16 slot within chunk
#pragma unroll
    for (int mt = 0; mt < 8; ++mt) {
#pragma unroll
      for (int r = 0; r < 8; ++r) {
        int row = mt * 16 + 8 * h + r;       // M = 8h + r, N = nl
        int chp = chx ^ ((row >> 1) & 7);
        sx[row * 64 + chp * 8 + low] = f2bf(acc[mt][r] + bcv);
      }
    }
  }
  __syncthreads();

  // --- linear: y = x @ Wl + bl ---
  Frag16 bl0, bl1;
  {
    const u32* p0 = wlf + ((0 * 4 + wv) * 32 + lane) * 8;
    const u32* p1 = wlf + ((1 * 4 + wv) * 32 + lane) * 8;
    bl0.q[0] = *(const uint4*)(p0);
    bl0.q[1] = *(const uint4*)(p0 + 4);
    bl1.q[0] = *(const uint4*)(p1);
    bl1.q[1] = *(const uint4*)(p1 + 4);
  }
  const float blv = bl[16 * wv + nl];
#pragma unroll
  for (int mt = 0; mt < 8; ++mt) {
    v8f acc2 = zero8();
#pragma unroll
    for (int s = 0; s < 2; ++s) {
      Frag16 af;
      const u32* ap0 = smem.post.x + nl * 32 + (((4 * s + 2 * h)     ^ swz) * 4);
      const u32* ap1 = smem.post.x + nl * 32 + (((4 * s + 2 * h + 1) ^ swz) * 4);
      af.q[0] = *(const uint4*)(ap0 + mt * 512);
      af.q[1] = *(const uint4*)(ap1 + mt * 512);
      acc2 = __builtin_amdgcn_wmma_f32_16x16x32_bf16(
          false, af.v, false, (s == 0 ? bl0.v : bl1.v), (short)0, acc2,
          false, false);
    }
#pragma unroll
    for (int r = 0; r < 8; ++r)
      smem.post.y[(mt * 16 + 8 * h + r) * Y_STRIDE + 16 * wv + nl] =
          acc2[r] + blv;
  }
  __syncthreads();

  // --- LayerNorm over channels + store ---
  {
    const int row = tid;
    const float* yr = smem.post.y + row * Y_STRIDE;
    float mu = 0.0f;
#pragma unroll
    for (int c = 0; c < COUT; ++c) mu += yr[c];
    mu *= (1.0f / 64.0f);
    float var = 0.0f;
#pragma unroll
    for (int c = 0; c < COUT; ++c) {
      float d = yr[c] - mu;
      var += d * d;
    }
    var *= (1.0f / 64.0f);
    const float inv = rsqrtf(var + 1e-5f);
    const int v = v0 + row;
    if (v < N_VOXELS) {
      float4* op = (float4*)(out + (long long)v * COUT);
#pragma unroll
      for (int q = 0; q < COUT / 4; ++q) {
        float4 y = ((const float4*)yr)[q];
        float4 g = ((const float4*)gamma)[q];
        float4 b = ((const float4*)beta)[q];
        float4 o;
        o.x = g.x * (y.x - mu) * inv + b.x;
        o.y = g.y * (y.y - mu) * inv + b.y;
        o.z = g.z * (y.z - mu) * inv + b.z;
        o.w = g.w * (y.w - mu) * inv + b.w;
        op[q] = o;
      }
    }
  }
}

extern "C" void kernel_launch(void* const* d_in, const int* in_sizes, int n_in,
                              void* d_out, int out_size, void* d_ws, size_t ws_size,
                              hipStream_t stream) {
  (void)in_sizes; (void)n_in; (void)out_size; (void)ws_size;
  const float* feats = (const float*)d_in[0];
  const int*   nb    = (const int*)  d_in[1];
  const float* Wc    = (const float*)d_in[2];
  const float* bc    = (const float*)d_in[3];
  const float* Wl    = (const float*)d_in[4];
  const float* bl    = (const float*)d_in[5];
  const float* gamma = (const float*)d_in[6];
  const float* beta  = (const float*)d_in[7];
  float* out = (float*)d_out;

  // workspace carve-up (ws fully overwritten by prep kernels each call)
  u32* feats_bf = (u32*)d_ws;                       // N*32 uints = 12.8 MB
  u32* wcf      = feats_bf + (size_t)N_VOXELS * 32; // 221 KB
  u32* wlf      = wcf + WCONV_FRAG_U;               // 8 KB

  const int n_pairs = N_VOXELS * (CIN / 2);
  cvt_feats_kernel<<<(n_pairs + 255) / 256, 256, 0, stream>>>(feats, feats_bf, n_pairs);
  cvt_wconv_kernel<<<(WCONV_FRAG_U + 255) / 256, 256, 0, stream>>>(Wc, wcf);
  cvt_wlin_kernel<<<(WLIN_FRAG_U + 255) / 256, 256, 0, stream>>>(Wl, wlf);

  dim3 grid((N_VOXELS + TILE_M - 1) / TILE_M);
  dim3 block(TILE_M);
  sparse_conv_lin_ln_kernel<<<grid, block, 0, stream>>>(
      feats_bf, nb, wcf, bc, wlf, bl, gamma, beta, out);
}
